// FNO_Double_83348135346641
// MI455X (gfx1250) — matrix operand order repaired and checked by reading, not compiled, not run on hardware
//
#include <hip/hip_runtime.h>
#include <math.h>

typedef __attribute__((ext_vector_type(16))) _Float16 v16h;
typedef __attribute__((ext_vector_type(8)))  float    v8f;
typedef __attribute__((ext_vector_type(4)))  unsigned u32x4;
typedef __attribute__((ext_vector_type(8)))  int      i32x8;
typedef __attribute__((ext_vector_type(4)))  int      i32x4;

#define NB    8
#define NC    96
#define NH    256
#define NW    256
#define HW    (NH*NW)
#define NMODE 42
#define MPAD  48
#define NGRP  6
#define EPSN  1e-5f
#define PI_F  3.14159265358979323846f

// ---- CDNA5 wave32 WMMA fragment index helpers (ISA 7.12.2 layouts) ----
__device__ __forceinline__ int kidxA(int e, int hi) { return ((e < 8) ? e : e + 8) + hi * 8; }
__device__ __forceinline__ int kidxB(int e, int hi) { return ((e < 8) ? e : e + 8) + hi * 8; }

__device__ __forceinline__ v8f wmma32(v16h a, v16h b, v8f c) {
  return __builtin_amdgcn_wmma_f32_16x16x32_f16(false, a, false, b, (short)0, c, false, false);
}

__device__ __forceinline__ float gelu_exact(float x) {
  return 0.5f * x * (1.0f + erff(x * 0.70710678118654752f));
}

// =====================================================================
// Input projection: X = gelu(conv1x1(concat(up, noise, grid)))
// =====================================================================
__global__ void k_in_proj(const float* __restrict__ up, const float* __restrict__ noise,
                          const float* __restrict__ w, const float* __restrict__ b,
                          float* __restrict__ X) {
  __shared__ float wl[NC * 6];
  __shared__ float bl[NC];
  for (int i = threadIdx.x; i < NC * 6; i += blockDim.x) wl[i] = w[i];
  for (int i = threadIdx.x; i < NC; i += blockDim.x) bl[i] = b[i];
  __syncthreads();
  int p = blockIdx.x * blockDim.x + threadIdx.x;   // 0 .. NB*HW-1
  int bi = p >> 16;
  int px = p & (HW - 1);
  int hh = px >> 8, wc = px & 255;
  float in[6];
  in[0] = up[(size_t)(bi * 3 + 0) * HW + px];
  in[1] = up[(size_t)(bi * 3 + 1) * HW + px];
  in[2] = up[(size_t)(bi * 3 + 2) * HW + px];
  in[3] = noise[(size_t)bi * HW + px];
  in[4] = -1.0f + 2.0f * (float)wc / 255.0f;   // xx
  in[5] = -1.0f + 2.0f * (float)hh / 255.0f;   // yy
  for (int o = 0; o < NC; o++) {
    float acc = bl[o];
#pragma unroll
    for (int i = 0; i < 6; i++) acc += wl[o * 6 + i] * in[i];
    X[(size_t)(bi * NC + o) * HW + px] = gelu_exact(acc);
  }
}

// =====================================================================
// DELTA = conv1x1(dense) + depthwise3x3 + biases  (WMMA 96x96 per 128px)
// =====================================================================
__global__ void k_dense_local(const float* __restrict__ X,
                              const float* __restrict__ dw, const float* __restrict__ db,
                              const float* __restrict__ lw, const float* __restrict__ lb,
                              float* __restrict__ DELTA) {
  __shared__ _Float16 xs[128 * NC];   // [px][ch]
  __shared__ _Float16 wl[NC * NC];    // [o][i]
  __shared__ float lwl[NC * 9];
  __shared__ float dbl_[NC], lbl_[NC];
  int idx = blockIdx.x;
  int wc = idx & 1;
  int hh = (idx >> 1) & 255;
  int bb = idx >> 9;
  int wbase = wc * 128;
  const float* xb = X + (size_t)bb * NC * HW;
  for (int i = threadIdx.x; i < NC * 128; i += 256) {
    int ch = i >> 7, px = i & 127;
    xs[px * NC + ch] = (_Float16)xb[(size_t)ch * HW + hh * NW + wbase + px];
  }
  for (int i = threadIdx.x; i < NC * NC; i += 256) wl[i] = (_Float16)dw[i];
  for (int i = threadIdx.x; i < NC * 9; i += 256) lwl[i] = lw[i];
  for (int i = threadIdx.x; i < NC; i += 256) { dbl_[i] = db[i]; lbl_[i] = lb[i]; }
  __syncthreads();
  int lane = threadIdx.x & 31, wv = threadIdx.x >> 5;
  int hi = lane >> 4, l15 = lane & 15;
  int mpx = wv * 16;
  v16h afr[3];
#pragma unroll
  for (int kc = 0; kc < 3; kc++) {
    v16h a;
#pragma unroll
    for (int e = 0; e < 16; e++) a[e] = xs[(mpx + l15) * NC + kc * 32 + kidxA(e, hi)];
    afr[kc] = a;
  }
  int px0 = mpx + hi * 8;   // this lane's 8 consecutive pixels (matches D rows r+8*hi)
  for (int nt = 0; nt < 6; nt++) {
    v8f acc = {};
#pragma unroll
    for (int kc = 0; kc < 3; kc++) {
      v16h bfr;
#pragma unroll
      for (int e = 0; e < 16; e++) bfr[e] = wl[(nt * 16 + l15) * NC + kc * 32 + kidxB(e, hi)];
      acc = wmma32(afr[kc], bfr, acc);
    }
    int o = nt * 16 + l15;
    // ---- branchless depthwise 3x3: 3x10 register window, clamp+select ----
    const float* chp = xb + (size_t)o * HW;
    float win[3][10];
#pragma unroll
    for (int dy = 0; dy < 3; dy++) {
      int h2 = hh + dy - 1;
      bool hok = (h2 >= 0) & (h2 <= 255);
      int h2c = hok ? h2 : 0;
      const float* rowp = chp + (size_t)h2c * NW;
#pragma unroll
      for (int dx = 0; dx < 10; dx++) {
        int w2 = wbase + px0 - 1 + dx;
        bool ok = hok & (w2 >= 0) & (w2 <= 255);
        int w2c = min(max(w2, 0), 255);
        float v = rowp[w2c];
        win[dy][dx] = ok ? v : 0.0f;
      }
    }
    float lw9[9];
#pragma unroll
    for (int k = 0; k < 9; k++) lw9[k] = lwl[o * 9 + k];
    float bias = dbl_[o] + lbl_[o];
#pragma unroll
    for (int r = 0; r < 8; r++) {
      float s = bias;
#pragma unroll
      for (int dy = 0; dy < 3; dy++)
#pragma unroll
        for (int dx = 0; dx < 3; dx++) s += lw9[dy * 3 + dx] * win[dy][r + dx];
      DELTA[(size_t)(bb * NC + o) * HW + hh * NW + wbase + px0 + r] = acc[r] + s;
    }
  }
}

// =====================================================================
// Forward truncated DFT along W: z[h,wm] = sum_w x[h,w] * e^{-i*2pi*w*wm/256}
// =====================================================================
__global__ void k_dft_w(const float* __restrict__ X, _Float16* __restrict__ Zr,
                        _Float16* __restrict__ Zi) {
  __shared__ _Float16 Tc[MPAD * NW], Ts[MPAD * NW];   // [wm][w]
  int bc = blockIdx.x >> 1;
  int hh = blockIdx.x & 1;
  for (int i = threadIdx.x; i < MPAD * NW; i += 256) {
    int wm = i >> 8, w = i & 255;
    float th = 2.0f * PI_F * (float)(w * wm) / 256.0f;
    float s, c; __sincosf(th, &s, &c);
    Tc[i] = (_Float16)c;
    Ts[i] = (_Float16)(-s);
  }
  __syncthreads();
  int lane = threadIdx.x & 31, wv = threadIdx.x >> 5;
  int hi = lane >> 4, l15 = lane & 15;
  int row = hh * 128 + wv * 16 + l15;
  const float* xrow = X + (size_t)bc * HW + (size_t)row * NW;
  for (int nt = 0; nt < 3; nt++) {
    v8f ar = {}, ai = {};
    for (int kc = 0; kc < 8; kc++) {
      if (kc < 7) __builtin_prefetch(xrow + (kc + 1) * 32, 0, 0);
      v16h a;
#pragma unroll
      for (int e = 0; e < 16; e++) a[e] = (_Float16)xrow[kc * 32 + kidxA(e, hi)];
      v16h bcv, bsv;
#pragma unroll
      for (int e = 0; e < 16; e++) {
        int K = kc * 32 + kidxB(e, hi);
        int wm = nt * 16 + l15;
        bcv[e] = Tc[wm * NW + K];
        bsv[e] = Ts[wm * NW + K];
      }
      ar = wmma32(a, bcv, ar);
      ai = wmma32(a, bsv, ai);
    }
#pragma unroll
    for (int r = 0; r < 8; r++) {
      int rr = hh * 128 + wv * 16 + r + hi * 8;
      int wm = nt * 16 + l15;
      size_t off = ((size_t)bc * NH + rr) * MPAD + wm;
      Zr[off] = (_Float16)ar[r];
      Zi[off] = (_Float16)ai[r];
    }
  }
}

// =====================================================================
// Forward DFT along H: xf[hm,wm] = sum_h e^{-i*2pi*h*hm/256} * z[h,wm]
// =====================================================================
__global__ void k_dft_h(const _Float16* __restrict__ Zr, const _Float16* __restrict__ Zi,
                        _Float16* __restrict__ XFr, _Float16* __restrict__ XFi) {
  __shared__ _Float16 Ec[MPAD * NH], Es[MPAD * NH];   // [hm][h]
  int bc = blockIdx.x;
  for (int i = threadIdx.x; i < MPAD * NH; i += 256) {
    int hm = i >> 8, hh = i & 255;
    float th = 2.0f * PI_F * (float)(hh * hm) / 256.0f;
    float s, c; __sincosf(th, &s, &c);
    Ec[i] = (_Float16)c;
    Es[i] = (_Float16)s;
  }
  __syncthreads();
  int lane = threadIdx.x & 31, wv = threadIdx.x >> 5;
  int hi = lane >> 4, l15 = lane & 15;
  const _Float16* zr = Zr + (size_t)bc * NH * MPAD;
  const _Float16* zi = Zi + (size_t)bc * NH * MPAD;
  for (int t = wv; t < 9; t += 8) {
    int mt = t / 3, ntl = t % 3;
    v8f aCr = {}, aSr = {}, aCi = {}, aSi = {};
    for (int kc = 0; kc < 8; kc++) {
      v16h ac, as2;
#pragma unroll
      for (int e = 0; e < 16; e++) {
        int K = kc * 32 + kidxA(e, hi);
        int hm = mt * 16 + l15;
        ac[e] = Ec[hm * NH + K];
        as2[e] = Es[hm * NH + K];
      }
      v16h br, bi2;
#pragma unroll
      for (int e = 0; e < 16; e++) {
        int K = kc * 32 + kidxB(e, hi);
        int wm = ntl * 16 + l15;
        br[e] = zr[(size_t)K * MPAD + wm];
        bi2[e] = zi[(size_t)K * MPAD + wm];
      }
      aCr = wmma32(ac, br, aCr);
      aSr = wmma32(as2, br, aSr);
      aCi = wmma32(ac, bi2, aCi);
      aSi = wmma32(as2, bi2, aSi);
    }
#pragma unroll
    for (int r = 0; r < 8; r++) {
      int hm = mt * 16 + r + hi * 8;
      int wm = ntl * 16 + l15;
      size_t off = (size_t)bc * MPAD * MPAD + hm * MPAD + wm;
      XFr[off] = (_Float16)(aCr[r] + aSi[r]);   // re = cos*zr + sin*zi
      XFi[off] = (_Float16)(aCi[r] - aSr[r]);   // im = cos*zi - sin*zr
    }
  }
}

// =====================================================================
// Per-mode spectral mixing: of[b,o] = sum_i xf[b,i] * wc[i,o]  (complex)
// =====================================================================
__global__ void k_specmix(const _Float16* __restrict__ XFr, const _Float16* __restrict__ XFi,
                          const float* __restrict__ specw,
                          _Float16* __restrict__ OFr, _Float16* __restrict__ OFi) {
  int hm = blockIdx.x, wm = blockIdx.y;
  if (hm >= NMODE || wm >= NMODE) {
    for (int i = threadIdx.x; i < NB * NC; i += 256) {
      int b = i / NC, o = i % NC;
      size_t off = ((size_t)(b * NC + o) * MPAD + hm) * MPAD + wm;
      OFr[off] = (_Float16)0.f;
      OFi[off] = (_Float16)0.f;
    }
    return;
  }
  __shared__ _Float16 wr[NC * NC], wi[NC * NC];   // [o][i]
  __shared__ _Float16 ar[16 * NC], ai[16 * NC];   // [b][i]
  int m = hm * NMODE + wm;
  for (int i = threadIdx.x; i < NC * NC; i += 256) {
    int ii = i / NC, o = i % NC;
    size_t off = ((size_t)(ii * NC + o) * (NMODE * NMODE) + m) * 2;
    wr[o * NC + ii] = (_Float16)specw[off];
    wi[o * NC + ii] = (_Float16)specw[off + 1];
  }
  for (int i = threadIdx.x; i < 16 * NC; i += 256) {
    int b = i / NC, ii = i % NC;
    if (b < NB) {
      size_t off = ((size_t)(b * NC + ii) * MPAD + hm) * MPAD + wm;
      ar[i] = XFr[off];
      ai[i] = XFi[off];
    } else {
      ar[i] = (_Float16)0.f;
      ai[i] = (_Float16)0.f;
    }
  }
  __syncthreads();
  int lane = threadIdx.x & 31, wv = threadIdx.x >> 5;
  int hi = lane >> 4, l15 = lane & 15;
  if (wv >= 6) return;
  v8f RR = {}, II = {}, RI = {}, IR = {};
  for (int kc = 0; kc < 3; kc++) {
    v16h fr, fi;
#pragma unroll
    for (int e = 0; e < 16; e++) {
      int K = kc * 32 + kidxA(e, hi);
      fr[e] = ar[l15 * NC + K];
      fi[e] = ai[l15 * NC + K];
    }
    v16h brv, biv;
#pragma unroll
    for (int e = 0; e < 16; e++) {
      int K = kc * 32 + kidxB(e, hi);
      int o = wv * 16 + l15;
      brv[e] = wr[o * NC + K];
      biv[e] = wi[o * NC + K];
    }
    RR = wmma32(fr, brv, RR);
    II = wmma32(fi, biv, II);
    RI = wmma32(fr, biv, RI);
    IR = wmma32(fi, brv, IR);
  }
#pragma unroll
  for (int r = 0; r < 8; r++) {
    int b = r + hi * 8;
    if (b >= NB) continue;
    int o = wv * 16 + l15;
    size_t off = ((size_t)(b * NC + o) * MPAD + hm) * MPAD + wm;
    OFr[off] = (_Float16)(RR[r] - II[r]);
    OFi[off] = (_Float16)(RI[r] + IR[r]);
  }
}

// =====================================================================
// Inverse DFT along H: z[h,wm] = (1/256) sum_hm e^{+i*2pi*h*hm/256}*of[hm,wm]
// OF tile staged into LDS via CDNA5 TDM (tensor_load_to_lds) + per-lane
// global_load_async_to_lds_b32, overlapped with LDS twiddle generation.
// =====================================================================
__global__ void k_idft_h(const _Float16* __restrict__ OFr, const _Float16* __restrict__ OFi,
                         _Float16* __restrict__ Zr, _Float16* __restrict__ Zi) {
  __shared__ __align__(16) _Float16 Ac[NH * MPAD], As[NH * MPAD];  // [h][hm], scaled 1/256
  __shared__ __align__(16) _Float16 ofr_s[MPAD * MPAD];
  __shared__ __align__(16) _Float16 ofi_s[MPAD * MPAD];
  int bc = blockIdx.x;
  const _Float16* gofr = OFr + (size_t)bc * MPAD * MPAD;
  const _Float16* gofi = OFi + (size_t)bc * MPAD * MPAD;
  const unsigned tile_bytes = MPAD * MPAD * 2;   // 4608B per plane

#if __has_builtin(__builtin_amdgcn_tensor_load_to_lds)
  // --- OFr plane: Tensor Data Mover, 1D copy descriptor (D# groups 0/1) ---
  if (threadIdx.x == 0) {
    unsigned long long ga = (unsigned long long)(uintptr_t)gofr;
    u32x4 g0 = (u32x4)(0u);
    g0[0] = 1u;                                          // count=1, user mode
    g0[1] = (unsigned)(uintptr_t)(void*)ofr_s;           // lds_addr
    g0[2] = (unsigned)(ga & 0xFFFFFFFFu);                // global_addr[31:0]
    g0[3] = (unsigned)((ga >> 32) & 0x01FFFFFFu) | (2u << 30);  // addr[56:32] | type=2
    i32x8 g1 = (i32x8)(0);
    g1[0] = 0;                                           // wg_mask=0, data_size=1B
    g1[1] = (int)((tile_bytes & 0xFFFFu) << 16);         // tensor_dim0[15:0]
    g1[2] = (int)((tile_bytes >> 16) & 0xFFFFu) | (1 << 16);  // dim0 hi | tensor_dim1=1
    g1[3] = (int)((tile_bytes & 0xFFFFu) << 16);         // tile_dim0
    g1[5] = (int)tile_bytes;                             // tensor_dim0_stride lo
    i32x4 z4 = (i32x4)(0);
    i32x8 z8 = (i32x8)(0);
    __builtin_amdgcn_tensor_load_to_lds(g0, g1, z4, z4, z8, 0);
  }
#endif

  // --- OFi plane: per-lane async global->LDS copy (b32 granularity) ---
  {
    unsigned ldsbase = (unsigned)(uintptr_t)(void*)ofi_s;
    for (int i = threadIdx.x; i < (int)(tile_bytes / 4); i += 256) {
      unsigned goff = (unsigned)i * 4u;
      asm volatile("global_load_async_to_lds_b32 %0, %1, %2"
                   :: "v"(ldsbase + goff), "v"(goff), "s"(gofi)
                   : "memory");
    }
  }

  // --- overlap: generate twiddles while the async/TDM copies fly ---
  for (int i = threadIdx.x; i < NH * MPAD; i += 256) {
    int hh = i / MPAD, hm = i % MPAD;
    float th = 2.0f * PI_F * (float)(hh * hm) / 256.0f;
    float s, c; __sincosf(th, &s, &c);
    Ac[i] = (_Float16)(c * (1.0f / 256.0f));
    As[i] = (_Float16)(s * (1.0f / 256.0f));
  }

#if !__has_builtin(__builtin_amdgcn_tensor_load_to_lds)
  for (int i = threadIdx.x; i < MPAD * MPAD; i += 256) ofr_s[i] = gofr[i];
#endif

#if __has_builtin(__builtin_amdgcn_s_wait_asynccnt)
  __builtin_amdgcn_s_wait_asynccnt(0);
#else
  asm volatile("s_wait_asynccnt 0" ::: "memory");
#endif
#if __has_builtin(__builtin_amdgcn_s_wait_tensorcnt)
  __builtin_amdgcn_s_wait_tensorcnt(0);
#else
  asm volatile("s_wait_tensorcnt 0" ::: "memory");
#endif
  __syncthreads();

  int lane = threadIdx.x & 31, wv = threadIdx.x >> 5;
  int hi = lane >> 4, l15 = lane & 15;
  for (int t = wv; t < 48; t += 8) {
    int mt = t / 3, ntl = t % 3;
    v8f c1 = {}, c2 = {}, c3 = {}, c4 = {};
    for (int kc = 0; kc < 2; kc++) {
      v16h ac, as2;
#pragma unroll
      for (int e = 0; e < 16; e++) {
        int K = kc * 32 + kidxA(e, hi);
        int hh = mt * 16 + l15;
        _Float16 cv = (_Float16)0.f, sv = (_Float16)0.f;
        if (K < MPAD) { cv = Ac[hh * MPAD + K]; sv = As[hh * MPAD + K]; }
        ac[e] = cv; as2[e] = sv;
      }
      v16h br, bi2;
#pragma unroll
      for (int e = 0; e < 16; e++) {
        int K = kc * 32 + kidxB(e, hi);
        int wm = ntl * 16 + l15;
        _Float16 rv = (_Float16)0.f, iv = (_Float16)0.f;
        if (K < MPAD) { rv = ofr_s[K * MPAD + wm]; iv = ofi_s[K * MPAD + wm]; }
        br[e] = rv; bi2[e] = iv;
      }
      c1 = wmma32(ac, br, c1);    // cos*ofr
      c2 = wmma32(as2, bi2, c2);  // sin*ofi
      c3 = wmma32(ac, bi2, c3);   // cos*ofi
      c4 = wmma32(as2, br, c4);   // sin*ofr
    }
#pragma unroll
    for (int r = 0; r < 8; r++) {
      int hh = mt * 16 + r + hi * 8;
      int wm = ntl * 16 + l15;
      size_t off = ((size_t)bc * NH + hh) * MPAD + wm;
      Zr[off] = (_Float16)(c1[r] - c2[r]);
      Zi[off] = (_Float16)(c3[r] + c4[r]);
    }
  }
}

// =====================================================================
// Inverse rDFT along W (hermitian) + accumulate into DELTA
// y[h,w] = (1/256)*(zr[0] + 2*sum_{k=1..41}(zr*cos - zi*sin))
// =====================================================================
__global__ void k_idft_w_acc(const _Float16* __restrict__ Zr, const _Float16* __restrict__ Zi,
                             float* __restrict__ DELTA) {
  __shared__ _Float16 Bc[MPAD * NW], Bs[MPAD * NW];   // [wm][w] with hermitian coef
  int bc = blockIdx.x;
  for (int i = threadIdx.x; i < MPAD * NW; i += 256) {
    int wm = i >> 8, w = i & 255;
    float th = 2.0f * PI_F * (float)(w * wm) / 256.0f;
    float s, c; __sincosf(th, &s, &c);
    float coef = (wm == 0) ? (1.0f / 256.0f) : ((wm < NMODE) ? (2.0f / 256.0f) : 0.0f);
    Bc[i] = (_Float16)(c * coef);
    Bs[i] = (_Float16)(-s * coef);
  }
  __syncthreads();
  int lane = threadIdx.x & 31, wv = threadIdx.x >> 5;
  int hi = lane >> 4, l15 = lane & 15;
  const _Float16* zr = Zr + (size_t)bc * NH * MPAD;
  const _Float16* zi = Zi + (size_t)bc * NH * MPAD;
  float* dout = DELTA + (size_t)bc * HW;
  for (int t = wv; t < 256; t += 8) {
    int mt = t >> 4, ntl = t & 15;
    v8f acc = {};
    for (int kc = 0; kc < 2; kc++) {
      v16h azr, azi;
#pragma unroll
      for (int e = 0; e < 16; e++) {
        int K = kc * 32 + kidxA(e, hi);
        int hh = mt * 16 + l15;
        _Float16 rv = (_Float16)0.f, iv = (_Float16)0.f;
        if (K < MPAD) { rv = zr[(size_t)hh * MPAD + K]; iv = zi[(size_t)hh * MPAD + K]; }
        azr[e] = rv; azi[e] = iv;
      }
      v16h bcv, bsv;
#pragma unroll
      for (int e = 0; e < 16; e++) {
        int K = kc * 32 + kidxB(e, hi);
        int w = ntl * 16 + l15;
        _Float16 cv = (_Float16)0.f, sv = (_Float16)0.f;
        if (K < MPAD) { cv = Bc[K * NW + w]; sv = Bs[K * NW + w]; }
        bcv[e] = cv; bsv[e] = sv;
      }
      acc = wmma32(azr, bcv, acc);
      acc = wmma32(azi, bsv, acc);
    }
#pragma unroll
    for (int r = 0; r < 8; r++) {
      int hh = mt * 16 + r + hi * 8;
      int w = ntl * 16 + l15;
      dout[(size_t)hh * NW + w] += acc[r];
    }
  }
}

// =====================================================================
// GroupNorm pass 1: v = x + 0.5*delta (written back to X) + partial stats
// =====================================================================
__global__ void k_gn_partial(float* __restrict__ X, const float* __restrict__ DELTA,
                             float* __restrict__ P) {
  __shared__ float s1[256], s2[256];
  int g = blockIdx.x;       // 0..47 : b*6 + group
  int slice = blockIdx.y;   // 0..31
  int b = g / NGRP, grp = g % NGRP;
  size_t base = ((size_t)b * NC + grp * 16) * HW;
  int per = (16 * HW) / 32;
  float a1 = 0.f, a2 = 0.f;
  for (int i = slice * per + threadIdx.x; i < (slice + 1) * per; i += 256) {
    size_t off = base + i;
    float v = X[off] + 0.5f * DELTA[off];
    X[off] = v;
    a1 += v;
    a2 += v * v;
  }
  s1[threadIdx.x] = a1;
  s2[threadIdx.x] = a2;
  __syncthreads();
  for (int st = 128; st > 0; st >>= 1) {
    if (threadIdx.x < st) {
      s1[threadIdx.x] += s1[threadIdx.x + st];
      s2[threadIdx.x] += s2[threadIdx.x + st];
    }
    __syncthreads();
  }
  if (threadIdx.x == 0) {
    P[(g * 32 + slice) * 2 + 0] = s1[0];
    P[(g * 32 + slice) * 2 + 1] = s2[0];
  }
}

__global__ void k_gn_final(const float* __restrict__ P, float* __restrict__ G) {
  int g = threadIdx.x;
  if (g < 48) {
    float s1 = 0.f, s2 = 0.f;
    for (int i = 0; i < 32; i++) {
      s1 += P[(g * 32 + i) * 2 + 0];
      s2 += P[(g * 32 + i) * 2 + 1];
    }
    float n = 16.0f * (float)HW;
    float mean = s1 / n;
    float var = s2 / n - mean * mean;
    G[g * 2 + 0] = mean;
    G[g * 2 + 1] = rsqrtf(var + EPSN);
  }
}

__global__ void k_gn_apply(float* __restrict__ X, const float* __restrict__ G,
                           const float* __restrict__ gam, const float* __restrict__ bet) {
  size_t id = ((size_t)blockIdx.x * blockDim.x + threadIdx.x) * 4;
  if (id >= (size_t)NB * NC * HW) return;
  int chn = (int)(id >> 16);          // HW = 65536, 4-chunks never cross channels
  int c = chn % NC;
  int b = chn / NC;
  int g = b * NGRP + c / 16;
  float mean = G[g * 2], rstd = G[g * 2 + 1];
  float ga = gam[c], be = bet[c];
  float4 v = *(const float4*)(X + id);
  v.x = gelu_exact((v.x - mean) * rstd * ga + be);
  v.y = gelu_exact((v.y - mean) * rstd * ga + be);
  v.z = gelu_exact((v.z - mean) * rstd * ga + be);
  v.w = gelu_exact((v.w - mean) * rstd * ga + be);
  *(float4*)(X + id) = v;
}

// =====================================================================
// Output projection + residual + clip
// =====================================================================
__global__ void k_out_proj(const float* __restrict__ X, const float* __restrict__ up,
                           const float* __restrict__ w, const float* __restrict__ b,
                           float* __restrict__ out) {
  __shared__ float wl[3 * NC];
  __shared__ float bl[3];
  for (int i = threadIdx.x; i < 3 * NC; i += 256) wl[i] = w[i];
  if (threadIdx.x < 3) bl[threadIdx.x] = b[threadIdx.x];
  __syncthreads();
  int p = blockIdx.x * 256 + threadIdx.x;
  int bb = p >> 16, px = p & (HW - 1);
  float a0 = bl[0], a1 = bl[1], a2 = bl[2];
  for (int i = 0; i < NC; i++) {
    float xv = X[((size_t)bb * NC + i) * HW + px];
    a0 += wl[0 * NC + i] * xv;
    a1 += wl[1 * NC + i] * xv;
    a2 += wl[2 * NC + i] * xv;
  }
  float acc[3] = {a0, a1, a2};
#pragma unroll
  for (int o = 0; o < 3; o++) {
    float r = up[((size_t)bb * 3 + o) * HW + px] + acc[o];
    out[((size_t)bb * 3 + o) * HW + px] = fminf(fmaxf(r, 0.0f), 1.0f);
  }
}

// =====================================================================
extern "C" void kernel_launch(void* const* d_in, const int* in_sizes, int n_in,
                              void* d_out, int out_size, void* d_ws, size_t ws_size,
                              hipStream_t stream) {
  const float* up    = (const float*)d_in[0];
  const float* noise = (const float*)d_in[1];
  const float* ipw   = (const float*)d_in[2];
  const float* ipb   = (const float*)d_in[3];
  const float* specw = (const float*)d_in[4];
  const float* dw    = (const float*)d_in[5];
  const float* db    = (const float*)d_in[6];
  const float* lw    = (const float*)d_in[7];
  const float* lb    = (const float*)d_in[8];
  const float* ng    = (const float*)d_in[9];
  const float* nbp   = (const float*)d_in[10];
  const float* opw   = (const float*)d_in[11];
  const float* opb   = (const float*)d_in[12];

  char* wp = (char*)d_ws;
  auto take = [&](size_t bytes) {
    char* p = wp;
    wp += (bytes + 255) & ~(size_t)255;
    return p;
  };
  const size_t nact = (size_t)NB * NC * HW;             // 50,331,648
  float*    X     = (float*)take(nact * 4);
  float*    DELTA = (float*)take(nact * 4);
  _Float16* Zr    = (_Float16*)take((size_t)NB * NC * NH * MPAD * 2);
  _Float16* Zi    = (_Float16*)take((size_t)NB * NC * NH * MPAD * 2);
  _Float16* XFr   = (_Float16*)take((size_t)NB * NC * MPAD * MPAD * 2);
  _Float16* XFi   = (_Float16*)take((size_t)NB * NC * MPAD * MPAD * 2);
  _Float16* OFr   = (_Float16*)take((size_t)NB * NC * MPAD * MPAD * 2);
  _Float16* OFi   = (_Float16*)take((size_t)NB * NC * MPAD * MPAD * 2);
  float*    P     = (float*)take(48 * 32 * 2 * 4);
  float*    G     = (float*)take(48 * 2 * 4);

  k_in_proj<<<2048, 256, 0, stream>>>(up, noise, ipw, ipb, X);

  for (int l = 0; l < 4; l++) {
    k_dense_local<<<4096, 256, 0, stream>>>(X, dw + (size_t)l * NC * NC, db + l * NC,
                                            lw + (size_t)l * NC * 9, lb + l * NC, DELTA);
    k_dft_w<<<NB * NC * 2, 256, 0, stream>>>(X, Zr, Zi);
    k_dft_h<<<NB * NC, 256, 0, stream>>>(Zr, Zi, XFr, XFi);
    dim3 gsm(MPAD, MPAD);
    k_specmix<<<gsm, 256, 0, stream>>>(XFr, XFi, specw, OFr, OFi);
    k_idft_h<<<NB * NC, 256, 0, stream>>>(OFr, OFi, Zr, Zi);
    k_idft_w_acc<<<NB * NC, 256, 0, stream>>>(Zr, Zi, DELTA);
    dim3 gp(48, 32);
    k_gn_partial<<<gp, 256, 0, stream>>>(X, DELTA, P);
    k_gn_final<<<1, 64, 0, stream>>>(P, G);
    k_gn_apply<<<(unsigned)(nact / 1024), 256, 0, stream>>>(X, G, ng + l * NC, nbp + l * NC);
  }

  k_out_proj<<<2048, 256, 0, stream>>>(X, up, opw, opb, (float*)d_out);
}
